// Discriminator_minibatch_26362509263306
// MI455X (gfx1250) — matrix operand order repaired and checked by reading, not compiled, not run on hardware
//
#include <hip/hip_runtime.h>
#include <hip/hip_bf16.h>

typedef __attribute__((ext_vector_type(16))) _Float16 v16h;
typedef __attribute__((ext_vector_type(8)))  _Float16 v8h;
typedef __attribute__((ext_vector_type(4)))  _Float16 v4h;
typedef __attribute__((ext_vector_type(8)))  float    v8f;
typedef __attribute__((ext_vector_type(4)))  unsigned int v4u;
typedef __attribute__((ext_vector_type(8)))  unsigned int v8u;

// ---------------------------------------------------------------------------
// WMMA helpers (CDNA5 f16 16x16x32, f32 accumulate)
// ---------------------------------------------------------------------------
__device__ __forceinline__ v8f wmma16x16x32(v16h a, v16h b, v8f c) {
    return __builtin_amdgcn_wmma_f32_16x16x32_f16(
        /*neg_a=*/false, a, /*neg_b=*/false, b,
        /*c_mod=*/(short)0, c, /*reuse_a=*/false, /*reuse_b=*/false);
}

// A-fragment (16x32, f16) from row-major f16 matrix with leading dim lda.
// lane L: m = L&15, g = L>>4; halves j -> K = g*16 + j  (16 contiguous halves)
__device__ __forceinline__ v16h afrag_f16(const _Float16* A, int lda, int row0, int k0, int lane) {
    int m = lane & 15, g = lane >> 4;
    const _Float16* p = A + (size_t)(row0 + m) * lda + k0 + g * 16;
    union { v16h v; v8h h[2]; } u;
    u.h[0] = *(const v8h*)(p);
    u.h[1] = *(const v8h*)(p + 8);
    return u.v;
}

// B-fragment (32x16) where B[k][n] = W[n][k], W row-major f16 with leading dim ldw.
// lane L: n = L&15, kh = L>>4; halves j -> K = kh*16 + j  (16 contiguous halves)
__device__ __forceinline__ v16h bfrag(const _Float16* W, int ldw, int n0, int k0, int lane) {
    return *(const v16h*)(W + (size_t)(n0 + (lane & 15)) * ldw + k0 + (lane >> 4) * 16);
}

// D store to row-major f32: lane L: n=L&15, M = (L>>4)*8 + r
__device__ __forceinline__ void store_d_f32(float* C, int ldc, int row0, int col0, int lane, v8f d) {
    int n = lane & 15, mb = (lane >> 4) * 8;
#pragma unroll
    for (int r = 0; r < 8; ++r) C[(size_t)(row0 + mb + r) * ldc + col0 + n] = d[r];
}

__device__ __forceinline__ float sigmoidf_(float x) { return 1.f / (1.f + __expf(-x)); }

#define T_STEPS 32
#define NB      64
#define STATE   64
#define HID     128
#define ACT     32
#define CC      96
#define LDM     100   // padded row stride for sM (bank-conflict-free)
#define G3      384   // 3*HID

// ---------------------------------------------------------------------------
// K0: convert weights f32->f16; transpose Tm (128,12288) -> Tt (12288,128)
// ---------------------------------------------------------------------------
__global__ void __launch_bounds__(256)
k_convert(const float* __restrict__ wih0, const float* __restrict__ whh0,
          const float* __restrict__ wih1, const float* __restrict__ whh1,
          const float* __restrict__ w1,   const float* __restrict__ Tm,
          _Float16* __restrict__ wih0h, _Float16* __restrict__ whh0h,
          _Float16* __restrict__ wih1h, _Float16* __restrict__ whh1h,
          _Float16* __restrict__ w1h,   _Float16* __restrict__ Tth)
{
    size_t i0 = (size_t)blockIdx.x * blockDim.x + threadIdx.x;
    size_t stride = (size_t)gridDim.x * blockDim.x;
    for (size_t e = i0; e < (size_t)G3 * STATE; e += stride) wih0h[e] = (_Float16)wih0[e];
    for (size_t e = i0; e < (size_t)G3 * HID;   e += stride) whh0h[e] = (_Float16)whh0[e];
    for (size_t e = i0; e < (size_t)G3 * HID;   e += stride) wih1h[e] = (_Float16)wih1[e];
    for (size_t e = i0; e < (size_t)G3 * HID;   e += stride) whh1h[e] = (_Float16)whh1[e];
    for (size_t e = i0; e < (size_t)HID * (HID + ACT); e += stride) w1h[e] = (_Float16)w1[e];
    for (size_t e = i0; e < (size_t)HID * HID * CC; e += stride) {
        size_t q = e >> 7, k = e & 127;             // Tt[q][k] = Tflat[k][q]
        Tth[e] = (_Float16)Tm[k * (size_t)(HID * CC) + q];
    }
}

// ---------------------------------------------------------------------------
// K1: two-layer GRU, single workgroup (8 waves), sequential over T.
// Dynamic LDS: sGI f32 64x384 (96K) | sGHN f16 (16K) | sH0 f16 (16K) | sH1 f16 (16K) | sX f16 (8K)
// All WMMA loops have compile-time trip counts -> no EXEC predication.
// ---------------------------------------------------------------------------
__device__ __forceinline__ void gi_pass(const _Float16* sAh, const int Kin, const _Float16* W,
                                        const float* bias, float* sGI, int wave, int lane)
{
#pragma unroll 1
    for (int it = 0; it < 12; ++it) {               // 96 tiles / 8 waves
        int tile = wave + it * 8;
        int row0 = (tile & 3) * 16, col0 = (tile >> 2) * 16;
        float bv = bias[col0 + (lane & 15)];
        v8f acc = {bv, bv, bv, bv, bv, bv, bv, bv};
        for (int k0 = 0; k0 < Kin; k0 += 32) {
            v16h af = afrag_f16(sAh, Kin, row0, k0, lane);
            v16h bf = bfrag(W, Kin, col0, k0, lane);
            acc = wmma16x16x32(af, bf, acc);
        }
        store_d_f32(sGI, G3, row0, col0, lane, acc);
    }
}

// gh r/z part: tiles 0..63 (cols 0..255), accumulate onto gi in sGI
__device__ __forceinline__ void gh_rz_pass(const _Float16* sH, const _Float16* W, const float* bhh,
                                           float* sGI, int wave, int lane)
{
    int n = lane & 15, mb = (lane >> 4) * 8;
#pragma unroll 1
    for (int it = 0; it < 8; ++it) {
        int tile = wave + it * 8;
        int row0 = (tile & 3) * 16, col0 = (tile >> 2) * 16;
        float bv = bhh[col0 + n];
        v8f acc;
#pragma unroll
        for (int r = 0; r < 8; ++r) acc[r] = sGI[(size_t)(row0 + mb + r) * G3 + col0 + n] + bv;
#pragma unroll
        for (int k0 = 0; k0 < HID; k0 += 32) {
            v16h af = afrag_f16(sH, HID, row0, k0, lane);
            v16h bf = bfrag(W, HID, col0, k0, lane);
            acc = wmma16x16x32(af, bf, acc);
        }
#pragma unroll
        for (int r = 0; r < 8; ++r) sGI[(size_t)(row0 + mb + r) * G3 + col0 + n] = acc[r];
    }
}

// gh n part: tiles 64..95 (cols 256..383) -> sGHN (f16)
__device__ __forceinline__ void gh_n_pass(const _Float16* sH, const _Float16* W, const float* bhh,
                                          _Float16* sGHN, int wave, int lane)
{
    int n = lane & 15, mb = (lane >> 4) * 8;
#pragma unroll 1
    for (int it = 0; it < 4; ++it) {
        int tile = 64 + wave + it * 8;
        int row0 = (tile & 3) * 16, col0 = (tile >> 2) * 16;
        float bv = bhh[col0 + n];
        v8f acc = {bv, bv, bv, bv, bv, bv, bv, bv};
#pragma unroll
        for (int k0 = 0; k0 < HID; k0 += 32) {
            v16h af = afrag_f16(sH, HID, row0, k0, lane);
            v16h bf = bfrag(W, HID, col0, k0, lane);
            acc = wmma16x16x32(af, bf, acc);
        }
#pragma unroll
        for (int r = 0; r < 8; ++r)
            sGHN[(size_t)(row0 + mb + r) * HID + (col0 - 256) + n] = (_Float16)acc[r];
    }
}

__global__ void __launch_bounds__(256)
k_gru(const float* __restrict__ x,
      const _Float16* __restrict__ wih0h, const _Float16* __restrict__ whh0h,
      const _Float16* __restrict__ wih1h, const _Float16* __restrict__ whh1h,
      const float* __restrict__ bih0, const float* __restrict__ bhh0,
      const float* __restrict__ bih1, const float* __restrict__ bhh1,
      _Float16* __restrict__ ph)
{
    extern __shared__ char smem[];
    float*    sGI  = (float*)smem;                        // 64*384 f32   (96K)
    _Float16* sGHN = (_Float16*)(smem + 98304);           // 64*128 f16   (16K)
    _Float16* sH0  = (_Float16*)(smem + 98304 + 16384);   // 64*128 f16   (16K)
    _Float16* sH1  = (_Float16*)(smem + 98304 + 32768);   // 64*128 f16   (16K)
    _Float16* sX   = (_Float16*)(smem + 98304 + 49152);   // 64*64  f16   (8K)

    const int tid = threadIdx.x;
    const int lane = tid & 31, wave = tid >> 5;

    // f32 master copies of hidden states live in registers (32 per thread each)
    float h0r[32], h1r[32];
#pragma unroll
    for (int u = 0; u < 32; ++u) { h0r[u] = 0.f; h1r[u] = 0.f; }
    for (int e = tid; e < NB * HID; e += 256) { sH0[e] = (_Float16)0.f; sH1[e] = (_Float16)0.f; }
    __syncthreads();

#pragma unroll 1
    for (int t = 0; t < T_STEPS; ++t) {
        // stage x_t as f16 (vectorized: float4 -> v4h)
        const float* xt = x + (size_t)t * NB * STATE;
#pragma unroll
        for (int u = 0; u < 4; ++u) {
            int e = tid + u * 256;                 // 1024 chunks of 4
            float4 v = *(const float4*)(xt + e * 4);
            v4h hv = { (_Float16)v.x, (_Float16)v.y, (_Float16)v.z, (_Float16)v.w };
            *(v4h*)(sX + e * 4) = hv;
        }
        __syncthreads();

        // ---- layer 0 --------------------------------------------------------
        gi_pass(sX, STATE, wih0h, bih0, sGI, wave, lane);
        __syncthreads();
        gh_rz_pass(sH0, whh0h, bhh0, sGI, wave, lane);
        gh_n_pass (sH0, whh0h, bhh0, sGHN, wave, lane);
        __syncthreads();
#pragma unroll
        for (int u = 0; u < 32; ++u) {                 // gates -> h0
            int e = tid + u * 256;
            int i = e >> 7, c = e & 127;
            float r = sigmoidf_(sGI[(size_t)i * G3 + c]);
            float z = sigmoidf_(sGI[(size_t)i * G3 + HID + c]);
            float nn = tanhf(sGI[(size_t)i * G3 + 2 * HID + c] + r * (float)sGHN[e]);
            float h = (1.f - z) * nn + z * h0r[u];
            h0r[u] = h;
            sH0[e] = (_Float16)h;
        }
        __syncthreads();
        // ---- layer 1 --------------------------------------------------------
        gi_pass(sH0, HID, wih1h, bih1, sGI, wave, lane);
        __syncthreads();
        gh_rz_pass(sH1, whh1h, bhh1, sGI, wave, lane);
        gh_n_pass (sH1, whh1h, bhh1, sGHN, wave, lane);
        __syncthreads();
        _Float16* pt = ph + (size_t)t * NB * HID;
#pragma unroll
        for (int u = 0; u < 32; ++u) {                 // gates -> h1, emit p
            int e = tid + u * 256;
            int i = e >> 7, c = e & 127;
            float r = sigmoidf_(sGI[(size_t)i * G3 + c]);
            float z = sigmoidf_(sGI[(size_t)i * G3 + HID + c]);
            float nn = tanhf(sGI[(size_t)i * G3 + 2 * HID + c] + r * (float)sGHN[e]);
            float h = (1.f - z) * nn + z * h1r[u];
            h1r[u] = h;
            sH1[e] = (_Float16)h;
            pt[e]  = (_Float16)h;
        }
        __syncthreads();
    }
}

// ---------------------------------------------------------------------------
// K2: fc1 = relu(concat([p, a]) @ w1^T + b1)  -> f16 (32 blocks, one per t)
// ---------------------------------------------------------------------------
__global__ void __launch_bounds__(256)
k_fc1(const _Float16* __restrict__ ph, const float* __restrict__ a,
      const _Float16* __restrict__ w1h, const float* __restrict__ b1,
      _Float16* __restrict__ fc1h)
{
    __shared__ _Float16 sA[NB * (HID + ACT)];   // 64x160 f16 = 20KB
    const int t = blockIdx.x;
    const int tid = threadIdx.x, lane = tid & 31, wave = tid >> 5;

    // stage p (v8h chunks: 1024 chunks of 8 halves)
    const v8h* psrc = (const v8h*)(ph + (size_t)t * NB * HID);
#pragma unroll
    for (int u = 0; u < 4; ++u) {
        int e = tid + u * 256;
        *(v8h*)(sA + (e >> 4) * (HID + ACT) + (e & 15) * 8) = psrc[e];
    }
    // stage a (float4 -> v4h, 512 chunks of 4)
    const float* asrc = a + (size_t)t * NB * ACT;
#pragma unroll
    for (int u = 0; u < 2; ++u) {
        int e = tid + u * 256;
        float4 v = *(const float4*)(asrc + (e >> 3) * ACT + (e & 7) * 4);
        v4h hv = { (_Float16)v.x, (_Float16)v.y, (_Float16)v.z, (_Float16)v.w };
        *(v4h*)(sA + (e >> 3) * (HID + ACT) + HID + (e & 7) * 4) = hv;
    }
    __syncthreads();

#pragma unroll 1
    for (int it = 0; it < 4; ++it) {                  // 32 tiles / 8 waves
        int tile = wave + it * 8;
        int row0 = (tile & 3) * 16, col0 = (tile >> 2) * 16;
        float bv = b1[col0 + (lane & 15)];
        v8f acc = {bv, bv, bv, bv, bv, bv, bv, bv};
#pragma unroll
        for (int k0 = 0; k0 < HID + ACT; k0 += 32) {
            v16h af = afrag_f16(sA, HID + ACT, row0, k0, lane);
            v16h bf = bfrag(w1h, HID + ACT, col0, k0, lane);
            acc = wmma16x16x32(af, bf, acc);
        }
        int n = lane & 15, mb = (lane >> 4) * 8;
#pragma unroll
        for (int r = 0; r < 8; ++r) {
            float v = acc[r] > 0.f ? acc[r] : 0.f;
            fc1h[(size_t)t * NB * HID + (size_t)(row0 + mb + r) * HID + col0 + n] = (_Float16)v;
        }
    }
}

// ---------------------------------------------------------------------------
// K3: fused minibatch discrimination. Block (b, t):
//   stage fc1[t] (64x128 f16 tile) into LDS via Tensor Data Mover (TDM),
//   M = fc1[t] @ Tt[b]  (64x128 @ 128x96, WMMA) -> LDS (row stride 100)
//   norm[i][j] = sum_c |M[i][c]-M[j][c]| ; o_b[i] = (sum_j exp(-norm)-1)/63
// Dynamic LDS layout: sA f16 @0 (16K) | sM f32 (25.6K) | sACC (256B)
// ---------------------------------------------------------------------------
__global__ void __launch_bounds__(256)
k_discrim(const _Float16* __restrict__ fc1h, const _Float16* __restrict__ Tth,
          float* __restrict__ disc)
{
    extern __shared__ char dsm[];
    _Float16* sA   = (_Float16*)dsm;                  // offset 0 -> D#.lds_addr = 0
    float*    sM   = (float*)(dsm + 16384);
    float*    sACC = (float*)(dsm + 16384 + 25600);
    const int b = blockIdx.x, t = blockIdx.y;
    const int tid = threadIdx.x, lane = tid & 31, wave = tid >> 5;

    if (tid < NB) sACC[tid] = 0.f;

    // ---- TDM: DMA the 64x128 f16 fc1[t] tile into LDS (wave 0 issues) ------
    if (wave == 0) {
        unsigned long long ga = (unsigned long long)(const void*)(fc1h + (size_t)t * NB * HID);
        // D# group 0: count=1 | lds_addr=0 | global_addr | type=2
        v4u g0 = { 1u, 0u,
                   (unsigned)(ga & 0xffffffffull),
                   (unsigned)((ga >> 32) & 0x01ffffffull) | 0x80000000u };
        // D# group 1: data_size=2B; tensor_dim0=128; tensor_dim1=64;
        //             tile_dim0=128; tile_dim1=64; tensor_dim0_stride=128
        v8u g1 = { 0x10000u,          // workgroup_mask=0, data_size=1 (2 bytes)
                   128u << 16,        // [47:32]=0, [63:48]=tensor_dim0 lo
                   64u  << 16,        // [79:64]=tensor_dim0 hi=0, [95:80]=tensor_dim1 lo
                   128u << 16,        // [111:96]=tensor_dim1 hi=0, [127:112]=tile_dim0
                   64u,               // [143:128]=tile_dim1, [159:144]=tile_dim2=0
                   128u,              // [191:160]=tensor_dim0_stride lo
                   0u, 0u };
        asm volatile("tensor_load_to_lds %0, %1" :: "s"(g0), "s"(g1) : "memory");
        __builtin_amdgcn_s_wait_tensorcnt(0);
    }
    __syncthreads();

    // ---- GEMM: M = sA @ Tt[b]  (24 tiles / 8 waves) ------------------------
    const _Float16* Tb = Tth + (size_t)b * CC * HID;   // rows q=b*96+c, 128 k each
#pragma unroll 1
    for (int it = 0; it < 3; ++it) {
        int tile = wave + it * 8;
        int row0 = (tile & 3) * 16, col0 = (tile >> 2) * 16;
        v8f acc = {};
#pragma unroll
        for (int k0 = 0; k0 < HID; k0 += 32) {
            v16h af = afrag_f16(sA, HID, row0, k0, lane);
            v16h bf = bfrag(Tb, HID, col0, k0, lane);
            acc = wmma16x16x32(af, bf, acc);
        }
        store_d_f32(sM, LDM, row0, col0, lane, acc);
    }
    __syncthreads();

    // ---- pairwise L1 over C with register-cached Ri chunks ----------------
    const int i = tid >> 2, part = tid & 3;
    const float* Ri  = sM + i * LDM;
    const float* Rj0 = sM + (part * 16) * LDM;
    float nrm[16];
#pragma unroll
    for (int j = 0; j < 16; ++j) nrm[j] = 0.f;
#pragma unroll
    for (int ch = 0; ch < 3; ++ch) {                   // 3 chunks of 32 floats
        float4 xi[8];
#pragma unroll
        for (int c = 0; c < 8; ++c) xi[c] = *(const float4*)(Ri + ch * 32 + c * 4);
#pragma unroll
        for (int j = 0; j < 16; ++j) {
            const float* Rj = Rj0 + j * LDM + ch * 32;
            float s = 0.f;
#pragma unroll
            for (int c = 0; c < 8; ++c) {
                float4 xj = *(const float4*)(Rj + c * 4);
                s += fabsf(xi[c].x - xj.x) + fabsf(xi[c].y - xj.y) +
                     fabsf(xi[c].z - xj.z) + fabsf(xi[c].w - xj.w);
            }
            nrm[j] += s;
        }
    }
    float sum = 0.f;
#pragma unroll
    for (int j = 0; j < 16; ++j) sum += __expf(-nrm[j]);
    atomicAdd(&sACC[i], sum);
    __syncthreads();
    if (tid < NB)
        disc[((size_t)t * NB + tid) * HID + b] = (sACC[tid] - 1.0f) * (1.0f / 63.0f);
}

// ---------------------------------------------------------------------------
// K4: prob = sigmoid([fc1, o_b] @ w2^T + b2)
// ---------------------------------------------------------------------------
__global__ void __launch_bounds__(256)
k_final(const _Float16* __restrict__ fc1h, const float* __restrict__ disc,
        const float* __restrict__ w2, const float* __restrict__ b2,
        float* __restrict__ out)
{
    int idx = blockIdx.x * blockDim.x + threadIdx.x;   // (t*64 + i)
    if (idx >= T_STEPS * NB) return;
    const _Float16* f = fc1h + (size_t)idx * HID;
    const float*    d = disc + (size_t)idx * HID;
    float acc = b2[0];
#pragma unroll 8
    for (int c = 0; c < HID; ++c) acc += w2[c] * (float)f[c];
#pragma unroll 8
    for (int c = 0; c < HID; ++c) acc += w2[HID + c] * d[c];
    out[idx] = 1.f / (1.f + __expf(-acc));
}

// ---------------------------------------------------------------------------
// Host launcher
// ---------------------------------------------------------------------------
extern "C" void kernel_launch(void* const* d_in, const int* in_sizes, int n_in,
                              void* d_out, int out_size, void* d_ws, size_t ws_size,
                              hipStream_t stream)
{
    (void)in_sizes; (void)n_in; (void)out_size; (void)ws_size;
    const float* x    = (const float*)d_in[0];
    const float* a    = (const float*)d_in[1];
    const float* wih0 = (const float*)d_in[2];
    const float* whh0 = (const float*)d_in[3];
    const float* bih0 = (const float*)d_in[4];
    const float* bhh0 = (const float*)d_in[5];
    const float* wih1 = (const float*)d_in[6];
    const float* whh1 = (const float*)d_in[7];
    const float* bih1 = (const float*)d_in[8];
    const float* bhh1 = (const float*)d_in[9];
    const float* w1   = (const float*)d_in[10];
    const float* b1   = (const float*)d_in[11];
    const float* w2   = (const float*)d_in[12];
    const float* b2   = (const float*)d_in[13];
    const float* Tm   = (const float*)d_in[14];
    float* out = (float*)d_out;

    char* ws = (char*)d_ws;
    _Float16* wih0h = (_Float16*)(ws + 0);            //  49152 B
    _Float16* whh0h = (_Float16*)(ws + 49152);        //  98304 B
    _Float16* wih1h = (_Float16*)(ws + 147456);       //  98304 B
    _Float16* whh1h = (_Float16*)(ws + 245760);       //  98304 B
    _Float16* w1h   = (_Float16*)(ws + 344064);       //  40960 B
    _Float16* Tth   = (_Float16*)(ws + 385024);       // 3145728 B
    _Float16* ph    = (_Float16*)(ws + 3530752);      //  524288 B
    _Float16* fc1h  = (_Float16*)(ws + 4055040);      //  524288 B
    float*    disc  = (float*)   (ws + 4579328);      // 1048576 B  (total ~5.4 MB)

    k_convert<<<1024, 256, 0, stream>>>(wih0, whh0, wih1, whh1, w1, Tm,
                                        wih0h, whh0h, wih1h, whh1h, w1h, Tth);
    k_gru<<<1, 256, 155648, stream>>>(x, wih0h, whh0h, wih1h, whh1h,
                                      bih0, bhh0, bih1, bhh1, ph);
    k_fc1<<<32, 256, 0, stream>>>(ph, a, w1h, b1, fc1h);
    k_discrim<<<dim3(128, 32), 256, 42240, stream>>>(fc1h, Tth, disc);
    k_final<<<(T_STEPS * NB + 255) / 256, 256, 0, stream>>>(fc1h, disc, w2, b2, out);
}
